// GERL_55602646614234
// MI455X (gfx1250) — matrix operand order repaired and checked by reading, not compiled, not run on hardware
//
#include <hip/hip_runtime.h>

#define B_    128
#define C_    50
#define NN_   15
#define T_    30
#define D_    256
#define TD_   64
#define HID_  128
#define SEM_  320
#define DH_   32
#define NTOT  (B_*C_ + B_ + B_*NN_)   /* 8448 */

typedef __attribute__((ext_vector_type(16))) __bf16 v16bf;
typedef __attribute__((ext_vector_type(8)))  float  v8f;

struct __align__(16) U4 { unsigned int x, y, z, w; };
union Frag16 { U4 q[2]; unsigned int u[8]; v16bf v; };

__device__ __forceinline__ unsigned short f2bf(float f) {
  unsigned int u = __float_as_uint(f);
  u += 0x7fffu + ((u >> 16) & 1u);
  return (unsigned short)(u >> 16);
}
__device__ __forceinline__ float bf2f(unsigned short s) {
  return __uint_as_float(((unsigned int)s) << 16);
}
__device__ __forceinline__ v8f vzero() {
  v8f z = {0.f,0.f,0.f,0.f,0.f,0.f,0.f,0.f};
  return z;
}
__device__ __forceinline__ v8f wmma_bf16(const Frag16& a, const Frag16& b, v8f c) {
  return __builtin_amdgcn_wmma_f32_16x16x32_bf16(false, a.v, false, b.v, (short)0, c, false, false);
}

// A fragment: 16x32 bf16 tile at (row0, k0) of row-major [., ldm] bf16 matrix in LDS.
// Layout: lane&15 = M row; lanes 0-15 hold K = k0..k0+15, lanes 16-31 K = k0+16..k0+31
// (kh = lane>=16 ? 16 : 0) -> two contiguous b128 loads per lane.
__device__ __forceinline__ void loadA(Frag16& a, const unsigned short* mat, int ldm,
                                      int row0, int k0, int lane) {
  int m  = lane & 15;
  int kh = (lane & 16) ? 16 : 0;
  const unsigned short* p = mat + (row0 + m) * ldm + k0 + kh;
  a.q[0] = *(const U4*)(p);
  a.q[1] = *(const U4*)(p + 8);
}
// B fragment: 32x16 tile, from B^T stored row-major [N][ldm] (row n holds K values).
// Layout (same as loadA): lane&15 = N col; lanes 0-15 hold K = k0..k0+15, lanes 16-31 K = k0+16..k0+31.
__device__ __forceinline__ void loadBT(Frag16& b, const unsigned short* matT, int ldm,
                                       int n0, int k0, int lane) {
  int nn = lane & 15;
  int kb = k0 + ((lane & 16) ? 16 : 0);
  const unsigned short* p = matT + (n0 + nn) * ldm + kb;
  b.q[0] = *(const U4*)(p);
  b.q[1] = *(const U4*)(p + 8);
}
// B fragment from prepacked global weight buffer: [nt][kt][lane][8 dwords]
__device__ __forceinline__ void loadBW(Frag16& b, const unsigned int* WF, int KT,
                                       int nt, int kt, int lane) {
  const U4* p = (const U4*)(WF + (((nt*KT + kt)*32 + lane) << 3));
  b.q[0] = p[0];
  b.q[1] = p[1];
}

// ---- prepass: f32 [K][N] weight -> bf16 fragment layout ----
__global__ void wfrag_kernel(const float* __restrict__ W, unsigned int* __restrict__ out,
                             int KT, int N, int total) {
  int t = blockIdx.x * blockDim.x + threadIdx.x;
  if (t >= total) return;
  int v    = t & 7;
  int lane = (t >> 3) & 31;
  int tile = t >> 8;
  int kt = tile % KT;
  int nt = tile / KT;
  int n  = nt*16 + (lane & 15);
  int kk = kt*32 + ((lane & 16) ? 16 : 0) + 2*v;
  unsigned int lo = f2bf(W[kk*N + n]);
  unsigned int hi = f2bf(W[(kk+1)*N + n]);
  out[t] = lo | (hi << 16);
}

// ---- fused per-news title encoder ----
__global__ __launch_bounds__(256)
void tf_kernel(const int* __restrict__ clk_title, const int* __restrict__ clk_topic,
               const int* __restrict__ cand_title, const int* __restrict__ cand_topic,
               const int* __restrict__ nbr_title, const int* __restrict__ nbr_topic,
               const float* __restrict__ word_emb, const float* __restrict__ topic_emb,
               const unsigned int* __restrict__ wqF, const unsigned int* __restrict__ wkF,
               const unsigned int* __restrict__ wvF, const unsigned int* __restrict__ aW1F,
               const float* __restrict__ ab1, const float* __restrict__ aw2,
               float* __restrict__ sem) {
  __shared__ __align__(16) unsigned char smem[65536];
  unsigned short* qs = (unsigned short*)(smem);            // [32][256] Q
  unsigned short* ks = (unsigned short*)(smem + 16384);    // [32][256] K
  unsigned short* vt = (unsigned short*)(smem + 32768);    // [256][32] V transposed
  unsigned short* xb = (unsigned short*)(smem + 49152);    // [32][256] x (bf16)
  unsigned short* o_ = qs;                                 // [32][256] overlays Q after attn
  unsigned short* pb = xb;                                 // [8][32][32] probs, overlays x
  float* hsum = (float*)(smem + 16384);                    // [32][128] overlays K
  float* pl   = (float*)(smem + 32768);                    // [32] overlays vt (after use)
  float* pa   = pl + 32;                                   // [32]

  int i = blockIdx.x;
  int t = threadIdx.x;
  int lane = t & 31;
  int wave = t >> 5;

  const int* toks; int topic;
  if (i < B_*C_)           { toks = clk_title + i*T_;                topic = clk_topic[i]; }
  else if (i < B_*C_ + B_) { int j = i - B_*C_; toks = cand_title + j*T_; topic = cand_topic[j]; }
  else                     { int j = i - (B_*C_ + B_); toks = nbr_title + j*T_; topic = nbr_topic[j]; }

  // Stage A: gather word embeddings -> bf16 LDS, rows >= T_ zero padded
  for (int r = 0; r < 32; ++r) {
    float val = 0.f;
    if (r < T_) val = word_emb[(size_t)toks[r]*D_ + t];
    xb[r*D_ + t] = f2bf(val);
  }
  __syncthreads();

  // Stage B: Q,K,V = x @ W (48 N-tiles over 8 waves), dual independent M-chains
  for (int j = wave; j < 48; j += 8) {
    int mat = j >> 4;
    int nt  = j & 15;
    const unsigned int* WF = (mat == 0) ? wqF : (mat == 1) ? wkF : wvF;
    v8f c0 = vzero(), c1 = vzero();
#pragma unroll
    for (int kt = 0; kt < 8; ++kt) {
      Frag16 a0, a1, b;
      loadBW(b, WF, 8, nt, kt, lane);
      loadA(a0, xb, D_, 0,  kt*32, lane);
      loadA(a1, xb, D_, 16, kt*32, lane);
      c0 = wmma_bf16(a0, b, c0);
      c1 = wmma_bf16(a1, b, c1);
    }
    int mrow = (lane & 16) ? 8 : 0;
    int col  = nt*16 + (lane & 15);
    if (mat == 0) {
#pragma unroll
      for (int v = 0; v < 8; ++v) {
        qs[(mrow + v)*D_ + col]      = f2bf(c0[v]);
        qs[(16 + mrow + v)*D_ + col] = f2bf(c1[v]);
      }
    } else if (mat == 1) {
#pragma unroll
      for (int v = 0; v < 8; ++v) {
        ks[(mrow + v)*D_ + col]      = f2bf(c0[v]);
        ks[(16 + mrow + v)*D_ + col] = f2bf(c1[v]);
      }
    } else {
#pragma unroll
      for (int v = 0; v < 8; ++v) {
        vt[col*32 + mrow + v]      = f2bf(c0[v]);
        vt[col*32 + 16 + mrow + v] = f2bf(c1[v]);
      }
    }
  }
  __syncthreads();

  // Stage C: scores (4 independent WMMAs) + register softmax (one head per wave)
  {
    int h = wave;
    Frag16 qa0, qa1, kb0, kb1;
    loadA (qa0, qs, D_, 0,  h*DH_, lane);
    loadA (qa1, qs, D_, 16, h*DH_, lane);
    loadBT(kb0, ks, D_, 0,  h*DH_, lane);
    loadBT(kb1, ks, D_, 16, h*DH_, lane);
    v8f sf[2][2];
    sf[0][0] = wmma_bf16(qa0, kb0, vzero());
    sf[0][1] = wmma_bf16(qa0, kb1, vzero());
    sf[1][0] = wmma_bf16(qa1, kb0, vzero());
    sf[1][1] = wmma_bf16(qa1, kb1, vzero());

    const float scale = 0.17677669529663687f;  // 1/sqrt(32)
    bool mcol1 = ((lane & 15) >= 14);          // key columns 30,31 masked
    for (int mt = 0; mt < 2; ++mt) {
#pragma unroll
      for (int v = 0; v < 8; ++v) {
        float s0 = sf[mt][0][v] * scale;
        float s1 = mcol1 ? -1e30f : sf[mt][1][v] * scale;
        float mx = fmaxf(s0, s1);
#pragma unroll
        for (int d = 1; d < 16; d <<= 1) mx = fmaxf(mx, __shfl_xor(mx, d, 32));
        float e0 = expf(s0 - mx), e1 = expf(s1 - mx);
        float sm = e0 + e1;
#pragma unroll
        for (int d = 1; d < 16; d <<= 1) sm += __shfl_xor(sm, d, 32);
        float inv = 1.f / sm;
        int row = mt*16 + v + ((lane & 16) ? 8 : 0);
        pb[(h*32 + row)*32 + (lane & 15)]      = f2bf(e0 * inv);
        pb[(h*32 + row)*32 + 16 + (lane & 15)] = f2bf(e1 * inv);
      }
    }

    // Stage D: o = P @ V (4 independent WMMAs; per-head LDS slices, no barrier needed)
    Frag16 pa0, pa1, vb0, vb1;
    loadA (pa0, pb + h*32*32, 32, 0,  0, lane);
    loadA (pa1, pb + h*32*32, 32, 16, 0, lane);
    loadBT(vb0, vt, 32, h*DH_,      0, lane);
    loadBT(vb1, vt, 32, h*DH_ + 16, 0, lane);
    v8f o00 = wmma_bf16(pa0, vb0, vzero());
    v8f o01 = wmma_bf16(pa0, vb1, vzero());
    v8f o10 = wmma_bf16(pa1, vb0, vzero());
    v8f o11 = wmma_bf16(pa1, vb1, vzero());
    int mrow = (lane & 16) ? 8 : 0;
    int col0 = h*DH_ + (lane & 15);
#pragma unroll
    for (int v = 0; v < 8; ++v) {
      o_[(mrow + v)*D_ + col0]           = f2bf(o00[v]);
      o_[(mrow + v)*D_ + col0 + 16]      = f2bf(o01[v]);
      o_[(16 + mrow + v)*D_ + col0]      = f2bf(o10[v]);
      o_[(16 + mrow + v)*D_ + col0 + 16] = f2bf(o11[v]);
    }
  }
  __syncthreads();

  // Stage E: pooling MLP hidden: tanh(o @ aW1 + ab1) * aw2 (dual M-chains)
  {
    int nt = wave;  // 128 cols -> 8 tiles
    v8f c0 = vzero(), c1 = vzero();
#pragma unroll
    for (int kt = 0; kt < 8; ++kt) {
      Frag16 a0, a1, b;
      loadBW(b, aW1F, 8, nt, kt, lane);
      loadA(a0, o_, D_, 0,  kt*32, lane);
      loadA(a1, o_, D_, 16, kt*32, lane);
      c0 = wmma_bf16(a0, b, c0);
      c1 = wmma_bf16(a1, b, c1);
    }
    int col  = nt*16 + (lane & 15);
    float bb = ab1[col];
    float ww = aw2[col];
    int mrow = (lane & 16) ? 8 : 0;
#pragma unroll
    for (int v = 0; v < 8; ++v) {
      hsum[(mrow + v)*HID_ + col]      = tanhf(c0[v] + bb) * ww;
      hsum[(16 + mrow + v)*HID_ + col] = tanhf(c1[v] + bb) * ww;
    }
  }
  __syncthreads();

  if (t < 32) {
    float s = 0.f;
    for (int j = 0; j < HID_; ++j) s += hsum[t*HID_ + j];
    pl[t] = s;
  }
  __syncthreads();
  if (t < 32) {
    float mx = pl[0];
    for (int m = 1; m < T_; ++m) mx = fmaxf(mx, pl[m]);
    float sm = 0.f;
    for (int m = 0; m < T_; ++m) sm += expf(pl[m] - mx);
    pa[t] = (t < T_) ? expf(pl[t] - mx) / sm : 0.f;
  }
  __syncthreads();

  // Stage F: title = pa @ o, concat topic embedding
  {
    float acc = 0.f;
    for (int m = 0; m < T_; ++m) acc += pa[m] * bf2f(o_[m*D_ + t]);
    sem[(size_t)i*SEM_ + t] = acc;
    if (t < TD_) sem[(size_t)i*SEM_ + D_ + t] = topic_emb[topic*TD_ + t];
  }
}

// ---- masked additive-attention aggregation ----
__global__ __launch_bounds__(128)
void agg_kernel(const float* __restrict__ xbase, const int* __restrict__ gidx, int rowoff,
                int n, int dim, const int* __restrict__ mask,
                const float* __restrict__ W1, const float* __restrict__ b1,
                const float* __restrict__ w2, const float* __restrict__ b2,
                float* __restrict__ out) {
  __shared__ float red[128];
  __shared__ float lg[64];
  __shared__ float av[64];
  int b = blockIdx.x;
  int t = threadIdx.x;
  for (int j = 0; j < n; ++j) {
    const float* x = gidx ? (xbase + (size_t)gidx[b*n + j]*dim)
                          : (xbase + (size_t)(rowoff + b*n + j)*dim);
    float s = 0.f;
    for (int d = 0; d < dim; ++d) s += x[d] * W1[d*HID_ + t];
    red[t] = tanhf(s + b1[t]) * w2[t];
    __syncthreads();
    for (int st = 64; st > 0; st >>= 1) {
      if (t < st) red[t] += red[t + st];
      __syncthreads();
    }
    if (t == 0) lg[j] = mask[b*n + j] ? (red[0] + b2[0]) : -1e30f;
    __syncthreads();
  }
  if (t == 0) {
    float mx = lg[0];
    for (int j = 1; j < n; ++j) mx = fmaxf(mx, lg[j]);
    float sm = 0.f;
    for (int j = 0; j < n; ++j) { av[j] = expf(lg[j] - mx); sm += av[j]; }
    float inv = 1.f / sm;
    for (int j = 0; j < n; ++j) av[j] *= inv;
  }
  __syncthreads();
  for (int d = t; d < dim; d += 128) {
    float acc = 0.f;
    for (int j = 0; j < n; ++j) {
      const float* x = gidx ? (xbase + (size_t)gidx[b*n + j]*dim)
                            : (xbase + (size_t)(rowoff + b*n + j)*dim);
      acc += av[j] * x[d];
    }
    out[(size_t)b*dim + d] = acc;
  }
}

// ---- final assembly ----
__global__ void user_repr_kernel(const float* __restrict__ user_sem, const float* __restrict__ user_emb,
                                 const int* __restrict__ user_id, const float* __restrict__ user_two,
                                 float* __restrict__ out) {
  int b = blockIdx.x, t = threadIdx.x;
  float v;
  if (t < SEM_)            v = user_sem[b*SEM_ + t];
  else if (t < SEM_ + 128) v = user_emb[(size_t)user_id[b]*128 + (t - SEM_)];
  else                     v = user_two[b*128 + (t - SEM_ - 128)];
  out[b*576 + t] = v;
}

__global__ __launch_bounds__(576)
void news_repr_kernel(const float* __restrict__ sem, const float* __restrict__ news_emb,
                      const int* __restrict__ cand_id, const float* __restrict__ agg_id,
                      const float* __restrict__ agg_sem,
                      const float* __restrict__ proj_W, const float* __restrict__ proj_b,
                      float* __restrict__ out) {
  __shared__ float cat[896];
  int b = blockIdx.x, t = threadIdx.x;
  for (int idx = t; idx < 896; idx += 576) {
    float v;
    if (idx < 320)      v = sem[(size_t)(B_*C_ + b)*SEM_ + idx];
    else if (idx < 448) v = news_emb[(size_t)cand_id[b]*128 + (idx - 320)];
    else if (idx < 576) v = agg_id[b*128 + (idx - 448)];
    else                v = agg_sem[b*SEM_ + (idx - 576)];
    cat[idx] = v;
  }
  __syncthreads();
  float acc = proj_b[t];
  for (int r = 0; r < 896; ++r) acc += cat[r] * proj_W[r*576 + t];
  out[b*576 + t] = acc;
}

extern "C" void kernel_launch(void* const* d_in, const int* in_sizes, int n_in,
                              void* d_out, int out_size, void* d_ws, size_t ws_size,
                              hipStream_t stream) {
  (void)in_sizes; (void)n_in; (void)out_size; (void)ws_size;
  const int*   user_id       = (const int*)d_in[0];
  const int*   cand_id       = (const int*)d_in[1];
  const int*   cand_title    = (const int*)d_in[2];
  const int*   cand_topic    = (const int*)d_in[3];
  const int*   clk_title     = (const int*)d_in[4];
  const int*   clk_topic     = (const int*)d_in[5];
  const int*   clk_mask      = (const int*)d_in[6];
  const int*   nbr_users     = (const int*)d_in[7];
  const int*   nbr_user_mask = (const int*)d_in[8];
  const int*   nbr_news      = (const int*)d_in[9];
  const int*   nbr_news_mask = (const int*)d_in[10];
  const int*   nbr_title     = (const int*)d_in[11];
  const int*   nbr_topic     = (const int*)d_in[12];
  const float* user_emb      = (const float*)d_in[13];
  const float* news_emb      = (const float*)d_in[14];
  const float* word_emb      = (const float*)d_in[15];
  const float* topic_emb     = (const float*)d_in[16];
  const float* Wq  = (const float*)d_in[17];
  const float* Wk  = (const float*)d_in[18];
  const float* Wv  = (const float*)d_in[19];
  const float* aW1 = (const float*)d_in[20];
  const float* ab1 = (const float*)d_in[21];
  const float* aw2 = (const float*)d_in[22];
  const float* clk_W1 = (const float*)d_in[23];
  const float* clk_b1 = (const float*)d_in[24];
  const float* clk_w2 = (const float*)d_in[25];
  const float* clk_b2 = (const float*)d_in[26];
  const float* uagg_W1 = (const float*)d_in[27];
  const float* uagg_b1 = (const float*)d_in[28];
  const float* uagg_w2 = (const float*)d_in[29];
  const float* uagg_b2 = (const float*)d_in[30];
  const float* nid_W1 = (const float*)d_in[31];
  const float* nid_b1 = (const float*)d_in[32];
  const float* nid_w2 = (const float*)d_in[33];
  const float* nid_b2 = (const float*)d_in[34];
  const float* nsem_W1 = (const float*)d_in[35];
  const float* nsem_b1 = (const float*)d_in[36];
  const float* nsem_w2 = (const float*)d_in[37];
  const float* nsem_b2 = (const float*)d_in[38];
  const float* proj_W = (const float*)d_in[39];
  const float* proj_b = (const float*)d_in[40];

  char* ws = (char*)d_ws;
  unsigned int* wqF  = (unsigned int*)(ws + 0);         // 16nt x 8kt x 256 dw = 128KB
  unsigned int* wkF  = (unsigned int*)(ws + 131072);
  unsigned int* wvF  = (unsigned int*)(ws + 262144);
  unsigned int* aW1F = (unsigned int*)(ws + 393216);    // 8nt x 8kt x 256 dw = 64KB
  float* sem      = (float*)(ws + 458752);              // [8448][320] f32
  float* user_sem = (float*)(ws + 11272192);            // [128][320]
  float* agg_sem  = (float*)(ws + 11436032);            // [128][320]
  float* user_two = (float*)(ws + 11599872);            // [128][128]
  float* agg_id   = (float*)(ws + 11665408);            // [128][128]

  // weight fragment conversion (bf16, per-lane fragment layout)
  wfrag_kernel<<<128, 256, 0, stream>>>(Wq, wqF, 8, 256, 32768);
  wfrag_kernel<<<128, 256, 0, stream>>>(Wk, wkF, 8, 256, 32768);
  wfrag_kernel<<<128, 256, 0, stream>>>(Wv, wvF, 8, 256, 32768);
  wfrag_kernel<<<64, 256, 0, stream>>>(aW1, aW1F, 8, 128, 16384);

  // fused title encoders: clicked (6400) + candidate (128) + neighbor (1920)
  tf_kernel<<<NTOT, 256, 0, stream>>>(clk_title, clk_topic, cand_title, cand_topic,
                                      nbr_title, nbr_topic, word_emb, topic_emb,
                                      wqF, wkF, wvF, aW1F, ab1, aw2, sem);

  // aggregations
  agg_kernel<<<B_, 128, 0, stream>>>(sem, nullptr, 0, C_, SEM_, clk_mask,
                                     clk_W1, clk_b1, clk_w2, clk_b2, user_sem);
  agg_kernel<<<B_, 128, 0, stream>>>(user_emb, nbr_users, 0, 15, 128, nbr_user_mask,
                                     uagg_W1, uagg_b1, uagg_w2, uagg_b2, user_two);
  agg_kernel<<<B_, 128, 0, stream>>>(news_emb, nbr_news, 0, NN_, 128, nbr_news_mask,
                                     nid_W1, nid_b1, nid_w2, nid_b2, agg_id);
  agg_kernel<<<B_, 128, 0, stream>>>(sem, nullptr, B_*C_ + B_, NN_, SEM_, nbr_news_mask,
                                     nsem_W1, nsem_b1, nsem_w2, nsem_b2, agg_sem);

  float* out_user = (float*)d_out;
  float* out_news = out_user + (size_t)B_*576;
  user_repr_kernel<<<B_, 576, 0, stream>>>(user_sem, user_emb, user_id, user_two, out_user);
  news_repr_kernel<<<B_, 576, 0, stream>>>(sem, news_emb, cand_id, agg_id, agg_sem,
                                           proj_W, proj_b, out_news);
}
